// DNDFTree_1262720385121
// MI455X (gfx1250) — compile-verified, matrix operand order checked by reading, not run on hardware
//
#include <hip/hip_runtime.h>

// ---------------- problem constants (match reference) ----------------
#define TREE_DEPTH 6
#define IN_F       1024
#define OUT_F      1024
#define BATCH      2048
#define N_LEAVES   64
#define N_NODES    63

#define M_TILE     32          // batch rows per block in the WMMA kernel

// ---------------- CDNA5 WMMA types ----------------
typedef __attribute__((ext_vector_type(8)))  __bf16   v8bf;
typedef __attribute__((ext_vector_type(16))) __bf16   v16bf;
typedef __attribute__((ext_vector_type(8)))  float    v8f;
typedef __attribute__((ext_vector_type(8)))  _Float16 v8h;

// Load a 16-element bf16 fragment: two contiguous 8-bf16 (16B) chunks at
// p and p+16, matching the 16-bit A/B WMMA lane layout (K 0..7 + 16..23 for
// lanes 0-15, K 8..15 + 24..31 for lanes 16-31; caller applies the +8 half).
__device__ __forceinline__ v16bf load_frag16(const __bf16* p) {
  v8bf lo = *(const v8bf*)(p);
  v8bf hi = *(const v8bf*)(p + 16);
  return __builtin_shufflevector(lo, hi, 0, 1, 2, 3, 4, 5, 6, 7,
                                          8, 9, 10, 11, 12, 13, 14, 15);
}

// ---------------- f32 -> bf16 (RNE) conversion, 4 elems/thread ----------------
__device__ __forceinline__ unsigned int f2bf(float f) {
  unsigned int u = __float_as_uint(f);
  return (u + 0x7FFFu + ((u >> 16) & 1u)) >> 16;
}

__global__ void cvt_bf16_kernel(const float4* __restrict__ src,
                                uint2* __restrict__ dst, size_t n4) {
  size_t i = (size_t)blockIdx.x * blockDim.x + threadIdx.x;
  size_t stride = (size_t)gridDim.x * blockDim.x;
  for (; i < n4; i += stride) {
    float4 f = src[i];
    uint2 o;
    o.x = f2bf(f.x) | (f2bf(f.y) << 16);
    o.y = f2bf(f.z) | (f2bf(f.w) << 16);
    dst[i] = o;
  }
}

// ---------------- decision probabilities: sigmoid(x @ Wd^T + bd) ----------------
__global__ void decision_kernel(const float* __restrict__ x,
                                const float* __restrict__ Wd,
                                const float* __restrict__ bd,
                                float* __restrict__ dprob) {
  int t = blockIdx.x * blockDim.x + threadIdx.x;
  if (t >= BATCH * N_NODES) return;
  int b = t / N_NODES;
  int n = t - b * N_NODES;
  const float4* xr = (const float4*)(x + (size_t)b * IN_F);
  const float4* wr = (const float4*)(Wd + (size_t)n * IN_F);
  float s = 0.f;
  #pragma unroll 4
  for (int i = 0; i < IN_F / 4; ++i) {
    float4 a = xr[i], w = wr[i];
    s += a.x * w.x + a.y * w.y + a.z * w.z + a.w * w.w;
  }
  s += bd[n];
  dprob[(size_t)b * 64 + n] = 1.f / (1.f + __expf(-s));   // stride 64 for alignment
}

// ---------------- routing = prod over depth of branch probs ----------------
__global__ void routing_kernel(const float* __restrict__ dprob,
                               float* __restrict__ routing) {
  int t = blockIdx.x * blockDim.x + threadIdx.x;
  if (t >= BATCH * N_LEAVES) return;
  int b = t >> 6;
  int l = t & 63;
  int node = 0, index = l;
  float r = 1.f;
  #pragma unroll
  for (int d = 0; d < TREE_DEPTH; ++d) {
    float p = dprob[(size_t)b * 64 + node];
    r *= ((index & 1) == 0) ? p : (1.f - p);
    node = 2 * node + 1 + (index & 1);
    index >>= 1;
  }
  routing[t] = r;
}

// ---------------- zero-init output (harness poisons d_out) ----------------
__global__ void zero_kernel(float4* __restrict__ p, int n4) {
  int i = blockIdx.x * blockDim.x + threadIdx.x;
  if (i < n4) p[i] = float4{0.f, 0.f, 0.f, 0.f};
}

// ---------------- fused leaf GEMM (WMMA bf16) + softmax + routed accumulate ----
// grid: (BATCH/M_TILE, N_LEAVES), block: 256 threads (8 wave32 waves).
// Phase 1: async-copy the 32x1024 bf16 A tile (64KB) into LDS (ASYNCcnt path).
// Phase 2: K-outer / N-chunk-inner register-blocked WMMA: each wave owns 16
//          output columns in each of 8 column chunks -> 16 live 16x16 f32 acc
//          tiles (128 VGPRs), A fragments read once per K-step via ds_load.
// Phase 3: reuse the same 64KB LDS for f16 logits, block-wide softmax, and
//          atomic accumulation of routing[b,l]*softmax into d_out.
__global__ __launch_bounds__(256)
void leaf_softmax_kernel(const __bf16* __restrict__ xb,
                         const __bf16* __restrict__ Wlb,
                         const float* __restrict__ bl,
                         const float* __restrict__ routing,
                         float* __restrict__ out) {
  extern __shared__ char smem[];                 // 64KB, LDS offset 0
  __bf16*   sA   = (__bf16*)smem;                // phase 1-2: A tile [32][1024]
  _Float16* slog = (_Float16*)smem;              // phase 3: logits [32][1024]

  const int bm   = blockIdx.x * M_TILE;
  const int leaf = blockIdx.y;
  const int wave = threadIdx.x >> 5;
  const int lane = threadIdx.x & 31;
  const int lcol = lane & 15;
  const int half = (lane < 16) ? 0 : 8;          // K-half select for A/B frags
  const int rbase = (lane < 16) ? 0 : 8;         // C/D tile row half

  // ---- phase 1: async stage A tile (65536 bytes) into LDS ----
  {
    const char* gsrc = (const char*)(xb + (size_t)bm * IN_F);
    #pragma unroll
    for (int i = 0; i < 16; ++i) {
      unsigned ldsoff = (unsigned)threadIdx.x * 16u + (unsigned)i * 4096u;
      const char* g = gsrc + ldsoff;
      asm volatile("global_load_async_to_lds_b128 %0, %1, off"
                   :: "v"(ldsoff), "v"(g) : "memory");
    }
    asm volatile("s_wait_asynccnt 0x0" ::: "memory");
  }
  __syncthreads();

  // ---- phase 2: register-blocked WMMA GEMM over K ----
  const __bf16* Bbase = Wlb + (size_t)leaf * (size_t)OUT_F * IN_F;
  const __bf16* sA0 = sA + (size_t)lcol * IN_F;         // rows 0..15
  const __bf16* sA1 = sA + (size_t)(16 + lcol) * IN_F;  // rows 16..31

  v8f acc[8][2] = {};   // [n-chunk][m-tile], fully unrolled -> static regs

  #pragma unroll 1
  for (int k = 0; k < IN_F; k += 32) {
    v16bf a0 = load_frag16(sA0 + k + half);             // ds_load_b128 x2
    v16bf a1 = load_frag16(sA1 + k + half);
    #pragma unroll
    for (int c = 0; c < 8; ++c) {
      const __bf16* Bcol =
          Bbase + (size_t)(c * 128 + wave * 16 + lcol) * IN_F;
      v16bf bf = load_frag16(Bcol + k + half);          // global_load_b128 x2
      acc[c][0] = __builtin_amdgcn_wmma_f32_16x16x32_bf16(
          false, a0, false, bf, (short)0, acc[c][0], false, false);
      acc[c][1] = __builtin_amdgcn_wmma_f32_16x16x32_bf16(
          false, a1, false, bf, (short)0, acc[c][1], false, false);
    }
  }
  __syncthreads();      // all ds reads of sA done; LDS now reused for logits

  // ---- phase 3a: bias + store logits (f16) to LDS ----
  #pragma unroll
  for (int c = 0; c < 8; ++c) {
    const int ncol = c * 128 + wave * 16 + lcol;
    const float bias = bl[(size_t)leaf * OUT_F + ncol];
    #pragma unroll
    for (int j = 0; j < 8; ++j) {
      slog[(size_t)(rbase + j) * OUT_F + ncol]      = (_Float16)(acc[c][0][j] + bias);
      slog[(size_t)(16 + rbase + j) * OUT_F + ncol] = (_Float16)(acc[c][1][j] + bias);
    }
  }
  __syncthreads();

  // ---- phase 3b: softmax over OUT_F per row + routed atomic accumulate ----
  // 8 threads per row (contiguous lanes), 128 columns each.
  const int row = threadIdx.x >> 3;              // 0..31
  const int q   = threadIdx.x & 7;               // 0..7
  const v8h* lr = (const v8h*)(slog + (size_t)row * OUT_F + q * 128);

  float mx = -3.0e38f;
  #pragma unroll 4
  for (int c = 0; c < 16; ++c) {
    v8h h = lr[c];
    #pragma unroll
    for (int j = 0; j < 8; ++j) mx = fmaxf(mx, (float)h[j]);
  }
  mx = fmaxf(mx, __shfl_xor(mx, 1));
  mx = fmaxf(mx, __shfl_xor(mx, 2));
  mx = fmaxf(mx, __shfl_xor(mx, 4));

  float s = 0.f;
  #pragma unroll 4
  for (int c = 0; c < 16; ++c) {
    v8h h = lr[c];
    #pragma unroll
    for (int j = 0; j < 8; ++j) s += __expf((float)h[j] - mx);
  }
  s += __shfl_xor(s, 1);
  s += __shfl_xor(s, 2);
  s += __shfl_xor(s, 4);

  const float w = routing[(size_t)(bm + row) * N_LEAVES + leaf] / s;
  float* orow = out + (size_t)(bm + row) * OUT_F + q * 128;
  #pragma unroll 2
  for (int c = 0; c < 16; ++c) {
    v8h h = lr[c];
    #pragma unroll
    for (int j = 0; j < 8; ++j)
      atomicAdd(&orow[c * 8 + j], w * __expf((float)h[j] - mx));
  }
}

// ---------------- launch ----------------
extern "C" void kernel_launch(void* const* d_in, const int* in_sizes, int n_in,
                              void* d_out, int out_size, void* d_ws, size_t ws_size,
                              hipStream_t stream) {
  const float* x  = (const float*)d_in[0];
  const float* Wd = (const float*)d_in[1];
  const float* bd = (const float*)d_in[2];
  const float* Wl = (const float*)d_in[3];
  const float* bl = (const float*)d_in[4];
  float* out = (float*)d_out;

  // workspace layout
  constexpr size_t XB_ELEMS  = (size_t)BATCH * IN_F;                 // 2 Mi
  constexpr size_t WLB_ELEMS = (size_t)N_LEAVES * OUT_F * IN_F;      // 64 Mi
  char* ws = (char*)d_ws;
  unsigned short* xb_u  = (unsigned short*)ws;                       // 4 MB
  unsigned short* wlb_u = (unsigned short*)(ws + XB_ELEMS * 2);      // 128 MB
  float* dprob   = (float*)(ws + XB_ELEMS * 2 + WLB_ELEMS * 2);      // 512 KB
  float* routing = dprob + (size_t)BATCH * 64;                       // 512 KB

  // 1) bf16 conversions (pure bandwidth pass)
  cvt_bf16_kernel<<<2048, 256, 0, stream>>>(
      (const float4*)x, (uint2*)xb_u, XB_ELEMS / 4);
  cvt_bf16_kernel<<<8192, 256, 0, stream>>>(
      (const float4*)Wl, (uint2*)wlb_u, WLB_ELEMS / 4);

  // 2) decision probabilities + routing (tiny)
  decision_kernel<<<(BATCH * N_NODES + 255) / 256, 256, 0, stream>>>(
      x, Wd, bd, dprob);
  routing_kernel<<<(BATCH * N_LEAVES) / 256, 256, 0, stream>>>(
      dprob, routing);

  // 3) zero output, then fused WMMA leaf GEMM + softmax + routed accumulate
  zero_kernel<<<(out_size / 4 + 255) / 256, 256, 0, stream>>>(
      (float4*)out, out_size / 4);

  dim3 grid(BATCH / M_TILE, N_LEAVES);
  size_t lds_bytes = (size_t)M_TILE * OUT_F * sizeof(_Float16);      // 64 KB
  leaf_softmax_kernel<<<grid, 256, lds_bytes, stream>>>(
      (const __bf16*)xb_u, (const __bf16*)wlb_u, bl, routing, out);
}
